// BlockwiseQuantLinear_28046136443366
// MI455X (gfx1250) — 8-bit matrix operand order repaired and checked by reading, compiled, not run on hardware
//
#include <hip/hip_runtime.h>
#include <hip/hip_bf16.h>

#define M_DIM 8192
#define K_DIM 2048
#define N_DIM 2048
#define KBLK  16
#define FP8MAX 448.0f

#define BPITCH 144

typedef __attribute__((ext_vector_type(16))) int   v16i;
typedef __attribute__((ext_vector_type(8)))  float v8f;
typedef __attribute__((ext_vector_type(4)))  int   v4i;

#define AS1 __attribute__((address_space(1)))
#define AS3 __attribute__((address_space(3)))

#if __has_builtin(__builtin_amdgcn_global_load_async_to_lds_b128)
#define USE_ASYNC_LDS 1
#endif

static __device__ __forceinline__ void wait_async_all() {
#if defined(USE_ASYNC_LDS)
#if __has_builtin(__builtin_amdgcn_s_wait_asynccnt)
    __builtin_amdgcn_s_wait_asynccnt(0);
#else
    asm volatile("s_wait_asynccnt 0x0" ::: "memory");
#endif
#endif
}

__device__ __forceinline__ unsigned int f32_to_e4m3_sw(float f) {
    unsigned int u = __float_as_uint(f);
    unsigned int s = (u >> 31) << 7;
    float a = fabsf(f);
    if (!(a == a)) return s | 0x7Fu;
    if (a >= FP8MAX) return s | 0x7Eu;
    if (a == 0.0f) return s;
    int e = (int)((__float_as_uint(a) >> 23) & 0xFF) - 127;
    if (e < -6) e = -6;
    float step = exp2f((float)(e - 3));
    float q = rintf(a / step) * step;
    if (q == 0.0f) return s;
    if (q >= FP8MAX) return s | 0x7Eu;
    unsigned int qa = __float_as_uint(q);
    int qe = (int)((qa >> 23) & 0xFF) - 127;
    if (qe < -6) {
        unsigned int mant = (unsigned int)(q * 512.0f);
        return s | mant;
    }
    unsigned int mant = (qa >> 20) & 0x7u;
    return s | ((unsigned int)(qe + 7) << 3) | mant;
}

__global__ __launch_bounds__(256) void quant_kernel(const float* __restrict__ x,
                                                    unsigned char* __restrict__ xq,
                                                    float* __restrict__ xs) {
    const int lane = threadIdx.x & 31;
    const int gw   = blockIdx.x * 8 + (threadIdx.x >> 5);
    const int m    = gw >> 4;
    const int kb   = gw & (KBLK - 1);
    const size_t base = (size_t)m * K_DIM + (size_t)kb * 128;

    const float4 v = ((const float4*)(x + base))[lane];
    float am = fmaxf(fmaxf(fabsf(v.x), fabsf(v.y)), fmaxf(fabsf(v.z), fabsf(v.w)));
#pragma unroll
    for (int off = 16; off > 0; off >>= 1)
        am = fmaxf(am, __shfl_xor(am, off, 32));
    am = fmaxf(am, 1e-12f);
    const float scale = am * (1.0f / FP8MAX);
    const float inv   = FP8MAX / am;

    unsigned int packed;
#if __has_builtin(__builtin_amdgcn_cvt_pk_fp8_f32)
    packed = (unsigned int)__builtin_amdgcn_cvt_pk_fp8_f32(v.x * inv, v.y * inv, 0, false);
    packed = (unsigned int)__builtin_amdgcn_cvt_pk_fp8_f32(v.z * inv, v.w * inv, (int)packed, true);
#else
    packed =  f32_to_e4m3_sw(v.x * inv)
           | (f32_to_e4m3_sw(v.y * inv) << 8)
           | (f32_to_e4m3_sw(v.z * inv) << 16)
           | (f32_to_e4m3_sw(v.w * inv) << 24);
#endif
    ((unsigned int*)(xq + base))[lane] = packed;
    if (lane == 0) xs[(size_t)m * KBLK + kb] = scale;
}

__device__ __forceinline__ void stage_b(const unsigned char* __restrict__ wq,
                                        unsigned char* buf, int n0, int kb, int tid) {
    const int r_base = tid >> 3;
    const int c      = (tid & 7) * 16;
#pragma unroll
    for (int it = 0; it < 4; ++it) {
        const int r = it * 32 + r_base;
        const unsigned char* g = wq + (size_t)(n0 + r) * K_DIM + (size_t)kb * 128 + c;
        unsigned char* l = buf + r * BPITCH + c;
#if defined(USE_ASYNC_LDS)
        __builtin_amdgcn_global_load_async_to_lds_b128(
            (AS1 v4i*)(AS1 void*)(g), (AS3 v4i*)(AS3 void*)(l), 0, 0);
#else
        *(uint4*)l = *(const uint4*)g;
#endif
    }
}

__global__ __launch_bounds__(256) void gemm_kernel(const unsigned char* __restrict__ xq,
                                                   const float* __restrict__ xs,
                                                   const unsigned char* __restrict__ wq,
                                                   const float* __restrict__ ws,
                                                   float* __restrict__ y) {
    __shared__ unsigned char lds_b[2][128 * BPITCH];

    const int tid  = threadIdx.x;
    const int lane = tid & 31;
    const int wave = tid >> 5;
    const int wm = wave & 3;
    const int wn = wave >> 2;
    const int lm = lane & 15;
    const int h  = lane >> 4;

    const int m0 = blockIdx.y * 128 + wm * 32;
    const int n0 = blockIdx.x * 128 + wn * 64;
    const int nblk0 = blockIdx.x * 128;

    v8f acc[2][4];
#pragma unroll
    for (int t = 0; t < 2; ++t)
#pragma unroll
        for (int j = 0; j < 4; ++j)
#pragma unroll
            for (int i = 0; i < 8; ++i)
                acc[t][j][i] = 0.0f;

    stage_b(wq, &lds_b[0][0], nblk0, 0, tid);

    for (int kb = 0; kb < KBLK; ++kb) {
        wait_async_all();
        __syncthreads();
        if (kb + 1 < KBLK)
            stage_b(wq, &lds_b[(kb + 1) & 1][0], nblk0, kb + 1, tid);

        const float xs_l = xs[(size_t)(m0 + lane) * KBLK + kb];
        const float wsb  = ws[blockIdx.x * KBLK + kb];
        float sA[2][8];
#pragma unroll
        for (int t = 0; t < 2; ++t)
#pragma unroll
            for (int i = 0; i < 8; ++i)
                sA[t][i] = __shfl(xs_l, t * 16 + i + h * 8, 32) * wsb;

        v16i afr[2];
#pragma unroll
        for (int t = 0; t < 2; ++t) {
            const unsigned char* ap =
                xq + (size_t)(m0 + t * 16 + lm) * K_DIM + (size_t)kb * 128 + h * 8;
#pragma unroll
            for (int p = 0; p < 8; ++p) {
                const uint2 d = *(const uint2*)(ap + 8 * h + 32 * (p >> 1) + 8 * (p & 1));
                afr[t][2 * p]     = (int)d.x;
                afr[t][2 * p + 1] = (int)d.y;
            }
        }

        const unsigned char* bbuf = &lds_b[kb & 1][0];
#pragma unroll
        for (int j = 0; j < 4; ++j) {
            const unsigned char* bl = bbuf + (wn * 64 + 16 * j + lm) * BPITCH + h * 16;
            v16i bfr;
#pragma unroll
            for (int g = 0; g < 4; ++g) {
                const uint4 d = *(const uint4*)(bl + 32 * g);
                bfr[4 * g]     = (int)d.x;
                bfr[4 * g + 1] = (int)d.y;
                bfr[4 * g + 2] = (int)d.z;
                bfr[4 * g + 3] = (int)d.w;
            }
#pragma unroll
            for (int t = 0; t < 2; ++t) {
                v8f czero;
#pragma unroll
                for (int i = 0; i < 8; ++i) czero[i] = 0.0f;
                v8f part = __builtin_amdgcn_wmma_f32_16x16x128_fp8_fp8(
                    afr[t], bfr, (short)0, czero, false, false);
#pragma unroll
                for (int i = 0; i < 8; ++i)
                    acc[t][j][i] = fmaf(sA[t][i], part[i], acc[t][j][i]);
            }
        }
    }

#pragma unroll
    for (int t = 0; t < 2; ++t)
#pragma unroll
        for (int j = 0; j < 4; ++j) {
            const int col = n0 + 16 * j + lm;
#pragma unroll
            for (int i = 0; i < 8; ++i) {
                const int row = m0 + 16 * t + i + 8 * h;
                y[(size_t)row * N_DIM + col] = acc[t][j][i];
            }
        }
}

extern "C" void kernel_launch(void* const* d_in, const int* in_sizes, int n_in,
                              void* d_out, int out_size, void* d_ws, size_t ws_size,
                              hipStream_t stream) {
    const float*         x  = (const float*)d_in[0];
    const unsigned char* wq = (const unsigned char*)d_in[1];
    const float*         ws = (const float*)d_in[2];
    float*               y  = (float*)d_out;

    unsigned char* xq = (unsigned char*)d_ws;
    float*         xs = (float*)((char*)d_ws + (size_t)M_DIM * K_DIM);

    const int n_blocks = M_DIM * KBLK;
    quant_kernel<<<dim3(n_blocks / 8), dim3(256), 0, stream>>>(x, xq, xs);

    gemm_kernel<<<dim3(N_DIM / 128, M_DIM / 128), dim3(256), 0, stream>>>(xq, xs, wq, ws, y);
}
